// MaskedMultiHeadLatentRoPESelfAttention_7945689498262
// MI455X (gfx1250) — matrix operand order repaired and checked by reading, not compiled, not run on hardware
//
#include <hip/hip_runtime.h>
#include <hip/hip_bf16.h>

// ---------------------------------------------------------------------------
// MLA-style masked multi-head RoPE self-attention for MI455X (gfx1250).
// Heavy math on v_wmma_f32_16x16x32_bf16 (wave32). GEMM tiles are streamed by
// the Tensor Data Mover with LDS double-buffering (DMA overlaps WMMA);
// attention K tiles use global_load_async_to_lds_b128 (ASYNCcnt).
// ---------------------------------------------------------------------------

typedef __bf16 bf16_t;
typedef __attribute__((ext_vector_type(16))) __bf16 v16bf;
typedef __attribute__((ext_vector_type(8)))  __bf16 v8bf;
typedef __attribute__((ext_vector_type(8)))  float  v8f;
typedef __attribute__((ext_vector_type(4)))  unsigned int u32x4;
typedef __attribute__((ext_vector_type(8)))  unsigned int u32x8;

__device__ __forceinline__ v16bf cat8(v8bf lo, v8bf hi) {
  return __builtin_shufflevector(lo, hi, 0,1,2,3,4,5,6,7,8,9,10,11,12,13,14,15);
}

__device__ __forceinline__ v8f wmma_bf16(v16bf a, v16bf b, v8f c) {
  return __builtin_amdgcn_wmma_f32_16x16x32_bf16(false, a, false, b, (short)0, c,
                                                 false, false);
}

// --- CDNA5 Tensor Data Mover: 2D tile (bf16) global -> LDS ------------------
__device__ __forceinline__ void tdm_load_tile_2d(unsigned lds_off,
                                                 const void* gptr,
                                                 unsigned tile_w,  // elements
                                                 unsigned tile_h,  // rows
                                                 unsigned stride)  // elements
{
  unsigned long long ga = (unsigned long long)gptr;
  u32x4 g0;
  g0.x = 1u;                                   // count = 1 valid descriptor
  g0.y = lds_off;                              // LDS byte address
  g0.z = (unsigned)ga;                         // global_addr[31:0]
  g0.w = (unsigned)(ga >> 32) | (2u << 30);    // global_addr[56:32] | type=2
  u32x8 g1;
  g1.s0 = (1u << 16);                          // data_size = 2 bytes
  g1.s1 = (tile_w & 0xFFFFu) << 16;            // tensor_dim0[15:0]
  g1.s2 = (tile_w >> 16) | ((tile_h & 0xFFFFu) << 16); // dim0_hi | dim1_lo
  g1.s3 = (tile_h >> 16) | (tile_w << 16);     // dim1_hi | tile_dim0
  g1.s4 = (tile_h & 0xFFFFu);                  // tile_dim1 (tile_dim2 = 0)
  g1.s5 = stride;                              // tensor_dim0_stride[31:0]
  g1.s6 = 0u;
  g1.s7 = 0u;
  asm volatile("tensor_load_to_lds %0, %1" :: "s"(g0), "s"(g1) : "memory");
}

// --- CDNA5 async global -> LDS (per-lane), tracked on ASYNCcnt --------------
__device__ __forceinline__ void async_load_b128(unsigned lds_off,
                                                const void* gaddr) {
  asm volatile("global_load_async_to_lds_b128 %0, %1, off"
               :: "v"(lds_off), "v"(gaddr) : "memory");
}
__device__ __forceinline__ void wait_async0() {
  asm volatile("s_wait_asynccnt 0x0" ::: "memory");
}

// ---------------------------------------------------------------------------
// Elementwise converts
// ---------------------------------------------------------------------------
__global__ __launch_bounds__(256) void f32_to_bf16_kernel(
    const float* __restrict__ in, bf16_t* __restrict__ out, size_t n) {
  size_t i = (size_t)blockIdx.x * 256 + threadIdx.x;
  if (i < n) out[i] = (bf16_t)in[i];
}

// in: [K][N] f32 row-major  ->  out: [N][K] bf16 row-major
__global__ __launch_bounds__(256) void transpose_f32_to_bf16_kernel(
    const float* __restrict__ in, bf16_t* __restrict__ out, int K, int N) {
  size_t i = (size_t)blockIdx.x * 256 + threadIdx.x;
  if (i < (size_t)K * N) {
    int k = (int)(i / N);
    int n = (int)(i % N);
    out[(size_t)n * K + k] = (bf16_t)in[i];
  }
}

// ---------------------------------------------------------------------------
// GEMM: C[M,N](f32 or bf16) = A[M,K](bf16) * Bt[N,K](bf16)^T + bias[N]
// 64x64 tile per 128-thread block (4 waves), K-step 64 (8 WMMA / stage).
// TDM double-buffering: tile i+1 DMA overlaps tile i compute.
// Dynamic LDS: buf0 {As@0, Bs@8192}, buf1 {As@16384, Bs@24576}.
// ---------------------------------------------------------------------------
__global__ __launch_bounds__(128) void gemm_bf16_kernel(
    const bf16_t* __restrict__ A, const bf16_t* __restrict__ Bt,
    const float* __restrict__ bias, float* __restrict__ Cf,
    bf16_t* __restrict__ Cb, int M, int N, int K) {
  extern __shared__ char smem[];

  const int n0   = blockIdx.x * 64;
  const int m0   = blockIdx.y * 64;
  const int tid  = threadIdx.x;
  const int wave = tid >> 5;
  const int lane = tid & 31;
  const int lh   = lane & 15;
  const bool hi  = lane >= 16;
  const int off1 = hi ? 8 : 0;    // A-fragment K sub-offset
  const int off2 = hi ? 16 : 0;   // unused; B fragments also use off1

  v8f acc[4] = {};

  const int niter = K >> 6;  // K / 64
  // Prologue: DMA tile 0 into buffer 0.
  if (wave == 0) {
    tdm_load_tile_2d(0u,    &A[(size_t)m0 * K], 64u, 64u, (unsigned)K);
    tdm_load_tile_2d(8192u, &Bt[(size_t)n0 * K], 64u, 64u, (unsigned)K);
  }

  for (int it = 0; it < niter; ++it) {
    const int k0 = it * 64;
    const unsigned buf = (it & 1) ? 16384u : 0u;

    if (it + 1 < niter) {
      // Kick DMA of tile i+1 into the other buffer, then wait only for the
      // two older TDM loads (tile i) -> DMA overlaps the WMMA chain below.
      if (wave == 0) {
        const unsigned nb = ((it + 1) & 1) ? 16384u : 0u;
        tdm_load_tile_2d(nb, &A[(size_t)m0 * K + k0 + 64], 64u, 64u,
                         (unsigned)K);
        tdm_load_tile_2d(nb + 8192u, &Bt[(size_t)n0 * K + k0 + 64], 64u, 64u,
                         (unsigned)K);
      }
      if (it + 2 < niter) {  // L2 warm-up two tiles ahead
        __builtin_prefetch(&A[(size_t)(m0 + (tid >> 1)) * K + k0 + 128], 0, 3);
        __builtin_prefetch(&Bt[(size_t)(n0 + (tid >> 1)) * K + k0 + 128], 0, 3);
      }
      __builtin_amdgcn_s_wait_tensorcnt(2);
    } else {
      __builtin_amdgcn_s_wait_tensorcnt(0);
    }
    __syncthreads();

    const bf16_t* As = (const bf16_t*)(smem + buf);
    const bf16_t* Bs = (const bf16_t*)(smem + buf + 8192u);
    const int ar = wave * 16 + lh;

    // Hoist all fragments of this stage, then run 8 WMMAs back-to-back.
    v16bf a[2], b[2][4];
#pragma unroll
    for (int kc = 0; kc < 2; ++kc) {
      a[kc] = cat8(*(const v8bf*)&As[ar * 64 + kc * 32 + off1],
                   *(const v8bf*)&As[ar * 64 + kc * 32 + 16 + off1]);
#pragma unroll
      for (int j = 0; j < 4; ++j) {
        const int bc = j * 16 + lh;
        b[kc][j] = cat8(*(const v8bf*)&Bs[bc * 64 + kc * 32 + off1],
                        *(const v8bf*)&Bs[bc * 64 + kc * 32 + 16 + off1]);
      }
    }
#pragma unroll
    for (int kc = 0; kc < 2; ++kc)
#pragma unroll
      for (int j = 0; j < 4; ++j)
        acc[j] = wmma_bf16(a[kc], b[kc][j], acc[j]);

    __syncthreads();
  }

  // Epilogue: bias add, store (C layout: element e -> row e + 8*hi).
#pragma unroll
  for (int j = 0; j < 4; ++j) {
    const int col = n0 + j * 16 + lh;
    const float bv = bias ? bias[col] : 0.0f;
#pragma unroll
    for (int e = 0; e < 8; ++e) {
      const int row = m0 + wave * 16 + e + (hi ? 8 : 0);
      const float v = acc[j][e] + bv;
      if (Cb) Cb[(size_t)row * N + col] = (bf16_t)v;
      else    Cf[(size_t)row * N + col] = v;
    }
  }
}

// ---------------------------------------------------------------------------
// RoPE + repack: qproj[b,s,h,64], kvproj[b,s,(k|v) 2048] -> head-major
// q/k/v [b,h,s,64] bf16. q pre-scaled by 1/sqrt(64).
// ---------------------------------------------------------------------------
__global__ __launch_bounds__(256) void pack_rope_kernel(
    const bf16_t* __restrict__ qproj, const bf16_t* __restrict__ kvproj,
    bf16_t* __restrict__ Qo, bf16_t* __restrict__ Ko, bf16_t* __restrict__ Vo,
    int S) {
  size_t gid = (size_t)blockIdx.x * 256 + threadIdx.x;
  if (gid >= (size_t)2 * 16 * S * 32) return;
  const int i   = (int)(gid & 31);
  size_t bhs    = gid >> 5;
  const int s   = (int)(bhs % S);
  const int bhi = (int)(bhs / S);       // b*16 + h
  const int h   = bhi & 15;
  const int b   = bhi >> 4;

  // theta = 10000^(-i/32) = exp(-i * ln(10000)/32)
  const float theta = __expf(-(float)i * 0.28782313662425576f);
  const float ang   = (float)s * theta;
  float sn, cs;
  __sincosf(ang, &sn, &cs);

  const size_t rowq = ((size_t)b * S + s) * 1024 + h * 64 + 2 * i;
  const size_t rowk = ((size_t)b * S + s) * 2048 + h * 64 + 2 * i;
  const size_t dst  = ((size_t)bhi * S + s) * 64 + 2 * i;

  const float qs = 0.125f;  // 1/sqrt(DK=64) folded into q
  const float q1 = (float)qproj[rowq], q2 = (float)qproj[rowq + 1];
  Qo[dst]     = (bf16_t)((q1 * cs - q2 * sn) * qs);
  Qo[dst + 1] = (bf16_t)((q1 * sn + q2 * cs) * qs);

  const float k1 = (float)kvproj[rowk], k2 = (float)kvproj[rowk + 1];
  Ko[dst]     = (bf16_t)(k1 * cs - k2 * sn);
  Ko[dst + 1] = (bf16_t)(k1 * sn + k2 * cs);

  Vo[dst]     = kvproj[rowk + 1024];
  Vo[dst + 1] = kvproj[rowk + 1024 + 1];
}

// ---------------------------------------------------------------------------
// Causal flash attention over one (b,h): block = 64 query rows, 4 waves.
// K tile staged with global_load_async_to_lds_b128 (ASYNCcnt), V transposed
// into LDS for contiguous P@V B-fragments. Online softmax in f32.
// Dynamic LDS: Ks @0 (8KB), Vt @8192 (8KB), Ps @16384 (8KB).
// ---------------------------------------------------------------------------
__global__ __launch_bounds__(128) void flash_attn_kernel(
    const bf16_t* __restrict__ Q, const bf16_t* __restrict__ Kk,
    const bf16_t* __restrict__ V, bf16_t* __restrict__ Out, int S) {
  extern __shared__ char smem[];
  bf16_t* Ks = (bf16_t*)smem;              // [key][dim]   64x64
  bf16_t* Vt = (bf16_t*)(smem + 8192);     // [dim][key]   64x64
  bf16_t* Ps = (bf16_t*)(smem + 16384);    // [wave][16][64]

  const int qblk = blockIdx.x;
  const int h    = blockIdx.y;
  const int b    = blockIdx.z;
  const int bh   = b * 16 + h;
  const int tid  = threadIdx.x;
  const int wave = tid >> 5;
  const int lane = tid & 31;
  const int lh   = lane & 15;
  const bool hi  = lane >= 16;
  const int off1 = hi ? 8 : 0;
  const int off2 = hi ? 16 : 0;

  const size_t base = (size_t)bh * S * 64;
  const int q0 = qblk * 64;

  // This wave's Q fragments (16 rows x 64 dims = two K=32 chunks).
  v16bf qf[2];
  {
    const size_t qrow = base + (size_t)(q0 + wave * 16 + lh) * 64;
#pragma unroll
    for (int kc = 0; kc < 2; ++kc)
      qf[kc] = cat8(*(const v8bf*)&Q[qrow + kc * 32 + off1],
                    *(const v8bf*)&Q[qrow + kc * 32 + 16 + off1]);
  }

  float mrow[8], lrow[8];
#pragma unroll
  for (int e = 0; e < 8; ++e) { mrow[e] = -3.0e38f; lrow[e] = 0.0f; }
  v8f o[4] = {};

  for (int kb = 0; kb <= qblk; ++kb) {
    const int k0g = kb * 64;
    // K tile: async DMA straight into LDS (row-major, matches B-fragment).
    // V tile: load through VGPRs and transpose into Vt.
#pragma unroll
    for (int it = 0; it < 4; ++it) {
      int cc = tid + it * 128;
      int r  = cc >> 3;
      int s8 = (cc & 7) * 8;
      async_load_b128((unsigned)(r * 128 + s8 * 2),
                      &Kk[base + (size_t)(k0g + r) * 64 + s8]);
      v8bf vv = *(const v8bf*)&V[base + (size_t)(k0g + r) * 64 + s8];
#pragma unroll
      for (int u = 0; u < 8; ++u) Vt[(s8 + u) * 64 + r] = vv[u];
    }
    if (kb < qblk) {  // L2 warm-up for the next K/V tiles
      const int r = tid >> 1;
      const int c = (tid & 1) * 32;
      __builtin_prefetch(&Kk[base + (size_t)(k0g + 64 + r) * 64 + c], 0, 3);
      __builtin_prefetch(&V[base + (size_t)(k0g + 64 + r) * 64 + c], 0, 3);
    }
    wait_async0();
    __syncthreads();

    // ---- scores = Q @ K^T (q pre-scaled); hoist B-frags per K-chunk ----
    v8f sc[4] = {};
#pragma unroll
    for (int kc = 0; kc < 2; ++kc) {
      v16bf bb[4];
#pragma unroll
      for (int j = 0; j < 4; ++j) {
        const int key = j * 16 + lh;
        bb[j] = cat8(*(const v8bf*)&Ks[key * 64 + kc * 32 + off1],
                     *(const v8bf*)&Ks[key * 64 + kc * 32 + 16 + off1]);
      }
#pragma unroll
      for (int j = 0; j < 4; ++j) sc[j] = wmma_bf16(qf[kc], bb[j], sc[j]);
    }

    // ---- causal mask (diagonal block only) ----
    if (kb == qblk) {
#pragma unroll
      for (int j = 0; j < 4; ++j) {
        const int key = j * 16 + lh;
#pragma unroll
        for (int e = 0; e < 8; ++e) {
          const int row = wave * 16 + e + (hi ? 8 : 0);
          if (key > row) sc[j][e] = -3.0e38f;
        }
      }
    }

    // ---- online softmax update (per lane: 8 rows, dup across 16 lanes) ----
    float alpha[8];
#pragma unroll
    for (int e = 0; e < 8; ++e) {
      float v = fmaxf(fmaxf(sc[0][e], sc[1][e]), fmaxf(sc[2][e], sc[3][e]));
#pragma unroll
      for (int d = 1; d < 16; d <<= 1) v = fmaxf(v, __shfl_xor(v, d, 32));
      const float nm = fmaxf(mrow[e], v);
      alpha[e] = __expf(mrow[e] - nm);
      mrow[e]  = nm;
    }
#pragma unroll
    for (int e = 0; e < 8; ++e) {
      const int prow = e + (hi ? 8 : 0);
      float rs = 0.0f;
#pragma unroll
      for (int j = 0; j < 4; ++j) {
        const float p = __expf(sc[j][e] - mrow[e]);
        rs += p;
        Ps[(wave * 16 + prow) * 64 + j * 16 + lh] = (bf16_t)p;
      }
#pragma unroll
      for (int d = 1; d < 16; d <<= 1) rs += __shfl_xor(rs, d, 32);
      lrow[e] = lrow[e] * alpha[e] + rs;
      o[0][e] *= alpha[e];
      o[1][e] *= alpha[e];
      o[2][e] *= alpha[e];
      o[3][e] *= alpha[e];
    }

    // ---- O += P @ V (P reloaded from LDS in A-fragment layout) ----
#pragma unroll
    for (int kc = 0; kc < 2; ++kc) {
      v16bf pa = cat8(*(const v8bf*)&Ps[(wave * 16 + lh) * 64 + kc * 32 + off1],
                      *(const v8bf*)&Ps[(wave * 16 + lh) * 64 + kc * 32 + 16 + off1]);
      v16bf bb[4];
#pragma unroll
      for (int j = 0; j < 4; ++j) {
        const int dcol = j * 16 + lh;
        bb[j] = cat8(*(const v8bf*)&Vt[dcol * 64 + kc * 32 + off1],
                     *(const v8bf*)&Vt[dcol * 64 + kc * 32 + 16 + off1]);
      }
#pragma unroll
      for (int j = 0; j < 4; ++j) o[j] = wmma_bf16(pa, bb[j], o[j]);
    }
    __syncthreads();
  }

  // ---- write normalized output: [b, s, h*64 + d] bf16 ----
#pragma unroll
  for (int j = 0; j < 4; ++j) {
    const int dcol = j * 16 + lh;
#pragma unroll
    for (int e = 0; e < 8; ++e) {
      const int row = q0 + wave * 16 + e + (hi ? 8 : 0);
      const float val = o[j][e] / lrow[e];
      Out[((size_t)b * S + row) * 1024 + h * 64 + dcol] = (bf16_t)val;
    }
  }
}

// ---------------------------------------------------------------------------
// Host side
// ---------------------------------------------------------------------------
extern "C" void kernel_launch(void* const* d_in, const int* in_sizes, int n_in,
                              void* d_out, int out_size, void* d_ws,
                              size_t ws_size, hipStream_t stream) {
  const int B = 2, S = 2048, E = 1024, H = 16, DK = 64, DV = 64, P = 512;
  const int M = B * S;  // 4096

  char* w = (char*)d_ws;
  auto carve = [&](size_t bytes) {
    char* p = w;
    w += (bytes + 255) & ~(size_t)255;
    return p;
  };

  bf16_t* xb     = (bf16_t*)carve((size_t)M * E * 2);
  bf16_t* WqT    = (bf16_t*)carve((size_t)E * E * 2);            // [1024][1024]
  bf16_t* Wkv1T  = (bf16_t*)carve((size_t)P * E * 2);            // [512][1024]
  bf16_t* Wkv2T  = (bf16_t*)carve((size_t)2048 * P * 2);         // [2048][512]
  bf16_t* WoT    = (bf16_t*)carve((size_t)E * E * 2);            // [1024][1024]
  bf16_t* qproj  = (bf16_t*)carve((size_t)M * 1024 * 2);
  bf16_t* midp   = (bf16_t*)carve((size_t)M * P * 2);
  bf16_t* kvproj = (bf16_t*)carve((size_t)M * 2048 * 2);
  bf16_t* qbh    = (bf16_t*)carve((size_t)B * H * S * DK * 2);
  bf16_t* kbh    = (bf16_t*)carve((size_t)B * H * S * DK * 2);
  bf16_t* vbh    = (bf16_t*)carve((size_t)B * H * S * DV * 2);
  bf16_t* attn   = (bf16_t*)carve((size_t)M * 1024 * 2);

  const float* x    = (const float*)d_in[0];
  const float* Wq   = (const float*)d_in[1];
  const float* bq   = (const float*)d_in[2];
  const float* Wkv1 = (const float*)d_in[3];
  const float* bkv1 = (const float*)d_in[4];
  const float* Wkv2 = (const float*)d_in[5];
  const float* bkv2 = (const float*)d_in[6];
  const float* Wo   = (const float*)d_in[7];
  const float* bo   = (const float*)d_in[8];

  // 1) converts
  {
    size_t n = (size_t)M * E;
    f32_to_bf16_kernel<<<dim3((unsigned)((n + 255) / 256)), 256, 0, stream>>>(
        x, xb, n);
  }
  transpose_f32_to_bf16_kernel<<<dim3((E * E + 255) / 256), 256, 0, stream>>>(
      Wq, WqT, E, E);
  transpose_f32_to_bf16_kernel<<<dim3((E * P + 255) / 256), 256, 0, stream>>>(
      Wkv1, Wkv1T, E, P);
  transpose_f32_to_bf16_kernel<<<dim3((P * 2048 + 255) / 256), 256, 0,
                                 stream>>>(Wkv2, Wkv2T, P, 2048);
  transpose_f32_to_bf16_kernel<<<dim3((E * E + 255) / 256), 256, 0, stream>>>(
      Wo, WoT, E, E);

  // 2) projection GEMMs (bf16 WMMA, f32 accumulate, bias in f32)
  dim3 blk(128);
  const size_t gemm_lds = 32768;  // 2 x (As 8KB + Bs 8KB) double-buffer
  gemm_bf16_kernel<<<dim3(1024 / 64, M / 64), blk, gemm_lds, stream>>>(
      xb, WqT, bq, nullptr, qproj, M, 1024, 1024);
  gemm_bf16_kernel<<<dim3(512 / 64, M / 64), blk, gemm_lds, stream>>>(
      xb, Wkv1T, bkv1, nullptr, midp, M, 512, 1024);
  gemm_bf16_kernel<<<dim3(2048 / 64, M / 64), blk, gemm_lds, stream>>>(
      midp, Wkv2T, bkv2, nullptr, kvproj, M, 2048, 512);

  // 3) RoPE + head-major repack (+ 1/sqrt(DK) folded into q)
  pack_rope_kernel<<<dim3((B * H * S * 32) / 256), 256, 0, stream>>>(
      qproj, kvproj, qbh, kbh, vbh, S);

  // 4) causal flash attention
  flash_attn_kernel<<<dim3(S / 64, H, B), blk, 24576, stream>>>(qbh, kbh, vbh,
                                                                attn, S);

  // 5) output GEMM -> f32 d_out
  gemm_bf16_kernel<<<dim3(1024 / 64, M / 64), blk, gemm_lds, stream>>>(
      attn, WoT, bo, (float*)d_out, nullptr, M, 1024, 1024);
}
